// StripDirectionalRWKVMix2D_58660663328794
// MI455X (gfx1250) — matrix operand order repaired and checked by reading, not compiled, not run on hardware
//
#include <hip/hip_runtime.h>
#include <hip/hip_bf16.h>

// ---------------------------------------------------------------------------
// StripDirectionalRWKVMix2D for MI455X (gfx1250, wave32, WMMA)
//   B=4, C=128, H=W=256, T=256, gamma-group g=32, shift=1
//
// Pipeline (all on `stream`):
//   k0: convert fuse_w / gate_w to f16 in workspace
//   k1: LDS-tiled H<->W transpose of x (coalesced col-direction reads)
//   k2: per-(dir,seq) RWKV mix: cooperative q_shift+mix into LDS, fp32 WMMA
//       GEMMs (k,v,r,out) fed by pure ds_loads, in-register WKV scan,
//       writes fused activations f16 in (dir,seq,t,c) layout
//   k3: fuse+gate: LDS-staged A-tile, f16 WMMA over K=512,
//       BN + SiLU + sigmoid-gate epilogue
// ---------------------------------------------------------------------------

#define B_   4
#define C_   128
#define H_   256
#define W_   256
#define T_   256
#define LTP  132          // padded LDS row stride (floats) -> conflict-free
#define ATP  520          // padded LDS row stride (halves) for phase-2 A tile

typedef __attribute__((ext_vector_type(2)))  float    v2f;
typedef __attribute__((ext_vector_type(4)))  float    v4f;
typedef __attribute__((ext_vector_type(8)))  float    v8f;
typedef __attribute__((ext_vector_type(8)))  _Float16 v8h;
typedef __attribute__((ext_vector_type(16))) _Float16 v16h;

// ---------------------------------------------------------------------------
// k0: weight conversion to f16
// ---------------------------------------------------------------------------
__global__ __launch_bounds__(256)
void cvt_w16_kernel(const float* __restrict__ fw, const float* __restrict__ gw,
                    _Float16* __restrict__ fw16, _Float16* __restrict__ gw16)
{
    int i = blockIdx.x * 256 + threadIdx.x;          // 65536 elements each
    fw16[i] = (_Float16)fw[i];
    gw16[i] = (_Float16)gw[i];
}

// ---------------------------------------------------------------------------
// k1: xT[b][c][w][h] = x[b][c][h][w]   (classic 32x32 LDS tile transpose)
// ---------------------------------------------------------------------------
__global__ __launch_bounds__(256)
void transpose_hw_kernel(const float* __restrict__ x, float* __restrict__ xT)
{
    __shared__ float tile[32][33];
    const int bc = blockIdx.z;                       // b*C + c  (0..511)
    const int h0 = blockIdx.y << 5;
    const int w0 = blockIdx.x << 5;
    const int tx = threadIdx.x;                      // 0..31
    const int ty = threadIdx.y;                      // 0..7
    const float* src = x  + (size_t)bc * (H_ * W_);
    float*       dst = xT + (size_t)bc * (H_ * W_);
    #pragma unroll
    for (int i = 0; i < 32; i += 8)
        tile[ty + i][tx] = src[(size_t)(h0 + ty + i) * W_ + (w0 + tx)];
    __syncthreads();
    #pragma unroll
    for (int i = 0; i < 32; i += 8)
        dst[(size_t)(w0 + ty + i) * H_ + (h0 + tx)] = tile[tx][ty + i];
}

// ---------------------------------------------------------------------------
// k2: directional RWKV mix.  grid = (1024 sequences, 4 directions),
// block = 256 threads = 8 waves.  Wave w owns output channels [16w,16w+16).
//
// q_shift semantics (halo rows zero-filled at sequence boundaries):
//   rows (d<2):  grp0 -> tok[t-1], grp1 -> tok[t+1], grp2/3 -> 0
//   cols (d>=2): grp2 -> tok[t-1], grp3 -> tok[t+1], grp0/1 -> 0
// ---------------------------------------------------------------------------
__global__ __launch_bounds__(256)
void dir_mix_kernel(const float* __restrict__ x,     const float* __restrict__ xT,
                    const float* __restrict__ mix_k, const float* __restrict__ mix_v,
                    const float* __restrict__ mix_r, const float* __restrict__ decay,
                    const float* __restrict__ first,
                    const float* __restrict__ Wk,    const float* __restrict__ Wv,
                    const float* __restrict__ Wr,    const float* __restrict__ Wo,
                    _Float16* __restrict__ fused16)
{
    __shared__ float lt[18][LTP];      // 16-token tile + q_shift halo
    __shared__ float xk_s[16][LTP];    // mixed activations (computed once)
    __shared__ float xv_s[16][LTP];
    __shared__ float xr_s[16][LTP];
    __shared__ float kt_s[16][LTP];    // k, then (sr*wkv)@Wo^T result
    __shared__ float vt_s[16][LTP];
    __shared__ float st_s[16][LTP];    // sigmoid(r), then sr * wkv_out
    __shared__ float p_mk[C_], p_mv[C_], p_mr[C_], p_w[C_], p_u[C_];

    const int tid  = threadIdx.x;
    const int d    = blockIdx.y;
    const int s    = blockIdx.x;
    const int lane = tid & 31;
    const int wv   = tid >> 5;
    const int mrow = lane & 15;        // A-matrix M / B-matrix N lane index
    const int hi   = lane >> 4;        // lane half selects K pair / M+8 rows
    const int koff = hi << 1;          // fp32 WMMA: lanes16-31 hold K=2,3
    const int n    = (wv << 4) + mrow; // this lane's output channel
    const int rev  = (d & 1);

    if (tid < C_) {
        p_mk[tid] = mix_k[d * C_ + tid];
        p_mv[tid] = mix_v[d * C_ + tid];
        p_mr[tid] = mix_r[d * C_ + tid];
        p_w[tid]  = -__expf(decay[d * C_ + tid] * (1.0f / T_));
        p_u[tid]  = first[d * C_ + tid] * (1.0f / T_);
    }

    const float* src  = (d < 2) ? x : xT;
    const float* base = src + (size_t)(s >> 8) * (C_ * H_ * W_)
                            + (size_t)(s & 255) * W_;   // + c*65536 + t

    const float* wkRow = Wk + (size_t)d * C_ * C_ + (size_t)n * C_;
    const float* wvRow = Wv + (size_t)d * C_ * C_ + (size_t)n * C_;
    const float* wrRow = Wr + (size_t)d * C_ * C_ + (size_t)n * C_;
    const float* woRow = Wo + (size_t)d * C_ * C_ + (size_t)n * C_;

    _Float16* dstSeq = fused16 + (size_t)(d * 1024 + s) * (T_ * C_);

    // WKV scan carry, persistent across token tiles (threads 0..127).
    float aa = 0.0f, bb = 0.0f, pp = -1e38f;
    float w_c = 0.0f, u_c = 0.0f;
    __syncthreads();
    if (tid < C_) { w_c = p_w[tid]; u_c = p_u[tid]; }

    for (int tile = 0; tile < T_ / 16; ++tile) {
        const int t0 = tile << 4;

        // ---- load 16-token tile + halo (zero-fill outside [0,T)) --------
        for (int i = tid; i < 18 * C_; i += 256) {
            int tr = i % 18;
            int c  = i / 18;
            int t  = t0 - 1 + tr;
            float val = 0.0f;
            if (t >= 0 && t < T_) {
                int tt = rev ? (T_ - 1 - t) : t;
                val = base[(size_t)c * (H_ * W_) + tt];
            }
            lt[tr][c] = val;
        }
        __syncthreads();

        // ---- q_shift + mix, computed ONCE cooperatively -----------------
        for (int i = tid; i < 16 * C_; i += 256) {
            int r = i >> 7, c = i & 127;
            int   grp   = c >> 5;
            int   gsel  = (d < 2) ? grp : (grp ^ 2);
            int   delta = (gsel == 0) ? -1 : ((gsel == 1) ? 1 : 0);
            float on    = (gsel < 2) ? 1.0f : 0.0f;
            float tok   = lt[r + 1][c];
            float xx    = on * lt[r + 1 + delta][c];
            float mk = p_mk[c], mv = p_mv[c], mr = p_mr[c];
            xk_s[r][c] = mk * tok + (1.0f - mk) * xx;
            xv_s[r][c] = mv * tok + (1.0f - mv) * xx;
            xr_s[r][c] = mr * tok + (1.0f - mr) * xx;
        }
        __syncthreads();

        // ---- k / v / r matmuls (fp32 WMMA, A from LDS, back-to-back) ----
        {
            v8f acc = {0,0,0,0,0,0,0,0};
            #pragma unroll 4
            for (int kk = 0; kk < 32; ++kk) {
                int c0 = (kk << 2) + koff;
                v2f a = *(const v2f*)(&xk_s[mrow][c0]);
                v2f b = *(const v2f*)(wkRow + c0);
                acc = __builtin_amdgcn_wmma_f32_16x16x4_f32(
                          false, a, false, b, (short)0, acc, false, false);
            }
            #pragma unroll
            for (int r = 0; r < 8; ++r) kt_s[r + (hi << 3)][n] = acc[r];
        }
        {
            v8f acc = {0,0,0,0,0,0,0,0};
            #pragma unroll 4
            for (int kk = 0; kk < 32; ++kk) {
                int c0 = (kk << 2) + koff;
                v2f a = *(const v2f*)(&xv_s[mrow][c0]);
                v2f b = *(const v2f*)(wvRow + c0);
                acc = __builtin_amdgcn_wmma_f32_16x16x4_f32(
                          false, a, false, b, (short)0, acc, false, false);
            }
            #pragma unroll
            for (int r = 0; r < 8; ++r) vt_s[r + (hi << 3)][n] = acc[r];
        }
        {
            v8f acc = {0,0,0,0,0,0,0,0};
            #pragma unroll 4
            for (int kk = 0; kk < 32; ++kk) {
                int c0 = (kk << 2) + koff;
                v2f a = *(const v2f*)(&xr_s[mrow][c0]);
                v2f b = *(const v2f*)(wrRow + c0);
                acc = __builtin_amdgcn_wmma_f32_16x16x4_f32(
                          false, a, false, b, (short)0, acc, false, false);
            }
            #pragma unroll
            for (int r = 0; r < 8; ++r)
                st_s[r + (hi << 3)][n] = 1.0f / (1.0f + __expf(-acc[r]));
        }
        __syncthreads();

        // ---- numerically-stable WKV scan, 16 steps, channel = tid -------
        if (tid < C_) {
            #pragma unroll
            for (int i = 0; i < 16; ++i) {
                float kt  = kt_s[i][tid];
                float vt  = vt_s[i][tid];
                float wwv = u_c + kt;
                float p   = fmaxf(pp, wwv);
                float e1  = __expf(pp - p);
                float e2  = __expf(wwv - p);
                float o   = (e1 * aa + e2 * vt) / (e1 * bb + e2);
                st_s[i][tid] *= o;                         // sr * out
                float ww2 = pp + w_c;
                float p2  = fmaxf(ww2, kt);
                float f1  = __expf(ww2 - p2);
                float f2  = __expf(kt - p2);
                aa = f1 * aa + f2 * vt;
                bb = f1 * bb + f2;
                pp = p2;
            }
        }
        __syncthreads();

        // ---- o = (sr*wkv) @ Wo^T  -> kt_s (reused) ----------------------
        {
            v8f acc = {0,0,0,0,0,0,0,0};
            #pragma unroll 4
            for (int kk = 0; kk < 32; ++kk) {
                int c0 = (kk << 2) + koff;
                v2f a = *(const v2f*)(&st_s[mrow][c0]);
                v2f b = *(const v2f*)(woRow + c0);
                acc = __builtin_amdgcn_wmma_f32_16x16x4_f32(
                          false, a, false, b, (short)0, acc, false, false);
            }
            #pragma unroll
            for (int r = 0; r < 8; ++r) kt_s[r + (hi << 3)][n] = acc[r];
        }
        __syncthreads();

        // ---- convert + coalesced store to fused16 (dir,seq,t,c) ---------
        _Float16* dstT = dstSeq + (size_t)t0 * C_;
        for (int i = tid; i < 16 * C_; i += 256) {
            int r = i >> 7, c = i & 127;
            dstT[(size_t)r * C_ + c] = (_Float16)kt_s[r][c];
        }
        __syncthreads();          // store done before lt/kt_s overwritten
    }
}

// ---------------------------------------------------------------------------
// k3: fuse + gate.  grid = 16384 tiles of 16 pixels (same b,hh row),
// block = 256 threads = 8 waves; each wave does Y and G for 16 channels.
// A-tile (16 pixels x 512 K) staged once in LDS, shared by all 8 waves.
// ---------------------------------------------------------------------------
__global__ __launch_bounds__(256)
void fuse_gate_kernel(const _Float16* __restrict__ fused16,
                      const _Float16* __restrict__ fw16,
                      const _Float16* __restrict__ gw16,
                      const float* __restrict__ bn_scale,
                      const float* __restrict__ bn_bias,
                      const float* __restrict__ bn_mean,
                      const float* __restrict__ bn_var,
                      const float* __restrict__ gate_b,
                      float* __restrict__ out)
{
    __shared__ _Float16 At[16][ATP];           // padded: 260 dwords/row

    const int tid  = threadIdx.x;
    const int lane = tid & 31;
    const int wv   = tid >> 5;
    const int mrow = lane & 15;
    const int hi   = lane >> 4;
    const int n    = (wv << 4) + mrow;         // output channel 0..127

    const int pix0 = blockIdx.x << 4;          // linear pixel (b*H+hh)*W + ww
    const int ww0t = pix0 & 255;               // tile's first ww
    const int hh   = (pix0 >> 8) & 255;        // same for all 16 pixels
    const int b    = pix0 >> 16;

    // ---- cooperative gather of A tile: 64 (row,dir)-chunks of 128 halves
    #pragma unroll
    for (int pass = 0; pass < 4; ++pass) {
        int rc  = (pass << 4) + (tid >> 4);    // 0..63
        int row = rc >> 2;
        int dd  = rc & 3;
        int off = (tid & 15) << 3;             // 0..120, 8 halves each
        int wwp = ww0t + row;
        int sD  = (dd < 2) ? (b * H_ + hh) : (b * W_ + wwp);
        int tD  = (dd == 0) ? wwp
                : (dd == 1) ? (W_ - 1 - wwp)
                : (dd == 2) ? hh
                :             (H_ - 1 - hh);
        const _Float16* srcp = fused16
            + ((size_t)(dd * 1024 + sD) * T_ + tD) * C_ + off;
        *(v8h*)(&At[row][(dd << 7) + off]) = *(const v8h*)srcp;
    }
    __syncthreads();

    const int koA = hi << 4;                   // f16 A: lane-half K offset
    const int koB = hi << 4;                   // f16 B: lane-half K offset

    v8f accY = {0,0,0,0,0,0,0,0};
    v8f accG = {0,0,0,0,0,0,0,0};

    #pragma unroll 4
    for (int kk = 0; kk < 16; ++kk) {
        int kb = kk << 5;
        v8h alo = *(const v8h*)(&At[mrow][kb + koA]);
        v8h ahi = *(const v8h*)(&At[mrow][kb + 8 + koA]);
        v16h av = __builtin_shufflevector(alo, ahi,
                    0,1,2,3,4,5,6,7,8,9,10,11,12,13,14,15);
        v16h bf = *(const v16h*)(fw16 + (size_t)n * 512 + kb + koB);
        v16h bg = *(const v16h*)(gw16 + (size_t)n * 512 + kb + koB);
        accY = __builtin_amdgcn_wmma_f32_16x16x32_f16(
                   false, av, false, bf, (short)0, accY, false, false);
        accG = __builtin_amdgcn_wmma_f32_16x16x32_f16(
                   false, av, false, bg, (short)0, accG, false, false);
    }

    // ---- BN + SiLU + sigmoid gate epilogue ------------------------------
    float inv  = bn_scale[n] * rsqrtf(bn_var[n] + 1e-5f);
    float bias = bn_bias[n] - bn_mean[n] * inv;
    float gb   = gate_b[n];

    float vals[8];
    #pragma unroll
    for (int r = 0; r < 8; ++r) {
        float y  = accY[r] * inv + bias;
        float sy = y / (1.0f + __expf(-y));                    // SiLU
        float g  = 1.0f / (1.0f + __expf(-(accG[r] + gb)));
        vals[r]  = sy * g;
    }

    // lane's 8 results are 8 consecutive ww positions -> two b128 stores
    const int ww0 = ww0t + (hi << 3);
    float* op = out + (((size_t)(b * C_ + n) * H_ + hh) * W_ + ww0);
    v4f o0, o1;
    o0.x = vals[0]; o0.y = vals[1]; o0.z = vals[2]; o0.w = vals[3];
    o1.x = vals[4]; o1.y = vals[5]; o1.z = vals[6]; o1.w = vals[7];
    *(v4f*)(op)     = o0;
    *(v4f*)(op + 4) = o1;
}

// ---------------------------------------------------------------------------
// host launcher
// ---------------------------------------------------------------------------
extern "C" void kernel_launch(void* const* d_in, const int* in_sizes, int n_in,
                              void* d_out, int out_size, void* d_ws, size_t ws_size,
                              hipStream_t stream)
{
    (void)in_sizes; (void)n_in; (void)out_size; (void)ws_size;

    const float* x        = (const float*)d_in[0];
    const float* mix_k    = (const float*)d_in[1];
    const float* mix_v    = (const float*)d_in[2];
    const float* mix_r    = (const float*)d_in[3];
    const float* decay    = (const float*)d_in[4];
    const float* first    = (const float*)d_in[5];
    const float* Wk       = (const float*)d_in[6];
    const float* Wv       = (const float*)d_in[7];
    const float* Wr       = (const float*)d_in[8];
    const float* Wo       = (const float*)d_in[9];
    const float* fuse_w   = (const float*)d_in[10];
    const float* bn_scale = (const float*)d_in[11];
    const float* bn_bias  = (const float*)d_in[12];
    const float* bn_mean  = (const float*)d_in[13];
    const float* bn_var   = (const float*)d_in[14];
    const float* gate_w   = (const float*)d_in[15];
    const float* gate_b   = (const float*)d_in[16];
    float* out = (float*)d_out;

    // workspace carve-up
    const size_t XT_BYTES    = (size_t)B_ * C_ * H_ * W_ * sizeof(float);     // 128 MiB
    const size_t FUSED_BYTES = (size_t)4 * 1024 * T_ * C_ * sizeof(_Float16); // 256 MiB
    const size_t FW_BYTES    = (size_t)C_ * 4 * C_ * sizeof(_Float16);        // 128 KiB
    char* w = (char*)d_ws;
    float*     xT      = (float*)w;
    _Float16*  fused16 = (_Float16*)(w + XT_BYTES);
    _Float16*  fw16    = (_Float16*)(w + XT_BYTES + FUSED_BYTES);
    _Float16*  gw16    = (_Float16*)(w + XT_BYTES + FUSED_BYTES + FW_BYTES);

    cvt_w16_kernel<<<256, 256, 0, stream>>>(fuse_w, gate_w, fw16, gw16);

    transpose_hw_kernel<<<dim3(W_ / 32, H_ / 32, B_ * C_), dim3(32, 8), 0, stream>>>(x, xT);

    dir_mix_kernel<<<dim3(1024, 4), 256, 0, stream>>>(
        x, xT, mix_k, mix_v, mix_r, decay, first, Wk, Wv, Wr, Wo, fused16);

    fuse_gate_kernel<<<(B_ * H_ * W_) / 16, 256, 0, stream>>>(
        fused16, fw16, gw16, bn_scale, bn_bias, bn_mean, bn_var, gate_b, out);
}
